// Reason_Ans_InterBlocks_56916906606854
// MI455X (gfx1250) — matrix operand order repaired and checked by reading, not compiled, not run on hardware
//
#include <hip/hip_runtime.h>

// ---------------------------------------------------------------------------
// MI455X (gfx1250) implementation of the heterogeneous graph-attention ref.
// Dominant GEMM (X[16384x1024] @ W[h,t][1024x64], 24 pairs) runs on
// v_wmma_f32_16x16x32_bf16 with hi/lo bf16 split (3 WMMAs) for ~fp32 accuracy.
// Attention (coefs^T @ h_last per (h,n)) also uses bf16-split WMMA with an
// LDS-transposed coefficient chunk pipeline.
// ---------------------------------------------------------------------------

typedef __attribute__((ext_vector_type(16))) __bf16        v16bf;
typedef __attribute__((ext_vector_type(8)))  float         v8f;
typedef __attribute__((ext_vector_type(16))) unsigned short u16x16;

#define NEG_INF_F (-9.0e15f)

static __device__ __forceinline__ unsigned short f2bf(float f) {
  unsigned u = __float_as_uint(f);
  u = u + 0x7FFFu + ((u >> 16) & 1u);   // round-to-nearest-even
  return (unsigned short)(u >> 16);
}
static __device__ __forceinline__ float bf2f(unsigned short h) {
  return __uint_as_float(((unsigned)h) << 16);
}

// Load a v16bf A/B operand as two 16B chunks: *(p) and *(p + off_elems).
static __device__ __forceinline__ v16bf ld16(const unsigned short* p, int off_elems) {
  struct P { uint4 a, b; } q;
  q.a = *(const uint4*)(p);
  q.b = *(const uint4*)(p + off_elems);
  return __builtin_bit_cast(v16bf, q);
}

static __device__ __forceinline__ v8f wmma_bf16(v16bf a, v16bf b, v8f c) {
  // (neg_a, A, neg_b, B, c_mod, C, reuse_a, reuse_b)
  return __builtin_amdgcn_wmma_f32_16x16x32_bf16(false, a, false, b, (short)0, c,
                                                 false, false);
}

// ---------------------------------------------------------------------------
// Kernel 1: split X (16384x1024 f32) into bf16 hi + residual lo.
// ---------------------------------------------------------------------------
__global__ __launch_bounds__(256) void prep_x_kernel(
    const float* __restrict__ X, unsigned short* __restrict__ Xhi,
    unsigned short* __restrict__ Xlo) {
  size_t i = ((size_t)blockIdx.x * 256 + threadIdx.x) * 4;
  float4 x = *(const float4*)(X + i);
  unsigned short h0 = f2bf(x.x), h1 = f2bf(x.y), h2 = f2bf(x.z), h3 = f2bf(x.w);
  unsigned short l0 = f2bf(x.x - bf2f(h0)), l1 = f2bf(x.y - bf2f(h1));
  unsigned short l2 = f2bf(x.z - bf2f(h2)), l3 = f2bf(x.w - bf2f(h3));
  uint2 ph; ph.x = (unsigned)h0 | ((unsigned)h1 << 16); ph.y = (unsigned)h2 | ((unsigned)h3 << 16);
  uint2 pl; pl.x = (unsigned)l0 | ((unsigned)l1 << 16); pl.y = (unsigned)l2 | ((unsigned)l3 << 16);
  *(uint2*)(Xhi + i) = ph;
  *(uint2*)(Xlo + i) = pl;
}

// ---------------------------------------------------------------------------
// Kernel 2: transpose W (24 x 1024 x 64) -> Wt (24 x 64 x 1024), bf16 hi/lo.
// ---------------------------------------------------------------------------
__global__ __launch_bounds__(256) void prep_w_kernel(
    const float* __restrict__ W, unsigned short* __restrict__ Whi,
    unsigned short* __restrict__ Wlo) {
  int o = blockIdx.x * 256 + threadIdx.x;     // o = ((ht*64)+d)*1024 + k
  int k = o & 1023;
  int rest = o >> 10;
  int d = rest & 63;
  int ht = rest >> 6;
  float w = W[((size_t)ht * 1024 + k) * 64 + d];
  unsigned short hh = f2bf(w);
  Whi[o] = hh;
  Wlo[o] = f2bf(w - bf2f(hh));
}

// ---------------------------------------------------------------------------
// Kernel 3: gating MLP.  g = sigmoid(relu(q @ qW1) @ qW2);
// emit w1[ht][n][d] = g1*a1, w2[ht][n][d] = g2*a2.  One block per (h,t).
// ---------------------------------------------------------------------------
__global__ __launch_bounds__(256) void gate_kernel(
    const float* __restrict__ q, const float* __restrict__ qW1,
    const float* __restrict__ qW2, const float* __restrict__ a1,
    const float* __restrict__ a2, float* __restrict__ w1,
    float* __restrict__ w2) {
  __shared__ float hb[64 * 128];
  const int ht = blockIdx.x;
  const int tid = threadIdx.x;
  const int n = tid >> 2;
  const int mq = (tid & 3) * 32;

  float acc[32];
#pragma unroll
  for (int m = 0; m < 32; ++m) acc[m] = 0.f;
  const float* qr = q + (size_t)n * 1024;
  const float* w1base = qW1 + (size_t)ht * 1024 * 128 + mq;
  for (int k = 0; k < 1024; ++k) {
    float qv = qr[k];
    const float* wr = w1base + (size_t)k * 128;
#pragma unroll
    for (int m = 0; m < 32; m += 4) {
      float4 wv = *(const float4*)(wr + m);
      acc[m + 0] = fmaf(qv, wv.x, acc[m + 0]);
      acc[m + 1] = fmaf(qv, wv.y, acc[m + 1]);
      acc[m + 2] = fmaf(qv, wv.z, acc[m + 2]);
      acc[m + 3] = fmaf(qv, wv.w, acc[m + 3]);
    }
  }
#pragma unroll
  for (int m = 0; m < 32; ++m) hb[n * 128 + mq + m] = fmaxf(acc[m], 0.f);
  __syncthreads();

  float acc2[32];
#pragma unroll
  for (int m = 0; m < 32; ++m) acc2[m] = 0.f;
  const float* w2base = qW2 + (size_t)ht * 128 * 128 + mq;
  for (int k = 0; k < 128; ++k) {
    float hv = hb[n * 128 + k];
    const float* wr = w2base + (size_t)k * 128;
#pragma unroll
    for (int m = 0; m < 32; m += 4) {
      float4 wv = *(const float4*)(wr + m);
      acc2[m + 0] = fmaf(hv, wv.x, acc2[m + 0]);
      acc2[m + 1] = fmaf(hv, wv.y, acc2[m + 1]);
      acc2[m + 2] = fmaf(hv, wv.z, acc2[m + 2]);
      acc2[m + 3] = fmaf(hv, wv.w, acc2[m + 3]);
    }
  }
  const float* a1p = a1 + (size_t)ht * 64;
  const float* a2p = a2 + (size_t)ht * 64;
  size_t wb = ((size_t)ht * 64 + n) * 64;
#pragma unroll
  for (int m = 0; m < 32; ++m) {
    int mm = mq + m;
    float g = 1.f / (1.f + __expf(-acc2[m]));
    if (mm < 64) w1[wb + mm] = g * a1p[mm];
    else         w2[wb + mm - 64] = g * a2p[mm - 64];
  }
}

// ---------------------------------------------------------------------------
// Kernel 4: main WMMA GEMM.  h[h,t] = X @ W[h,t] (16384x1024 x 1024x64).
// Epilogue computes s1/s2 via shfl reductions; t==2 stores adj_mask * h.
// One wave = one 16x64 output tile; block = 8 waves = 128 rows; grid (128,24).
// ---------------------------------------------------------------------------
__global__ __launch_bounds__(256) void gemm_h_kernel(
    const unsigned short* __restrict__ Xhi, const unsigned short* __restrict__ Xlo,
    const unsigned short* __restrict__ Wthi, const unsigned short* __restrict__ Wtlo,
    const float* __restrict__ w1, const float* __restrict__ w2,
    const float* __restrict__ adj_mask,
    float* __restrict__ s1, float* __restrict__ s2, float* __restrict__ hmask) {
  const int ht = blockIdx.y;
  const int t = ht % 3;
  const int h = ht / 3;
  const int wave = threadIdx.x >> 5;
  const int lane = threadIdx.x & 31;
  const int l15 = lane & 15;
  const int hi16 = lane >> 4;
  const int row0 = blockIdx.x * 128 + wave * 16;  // global M row (= n*256 + e)
  const int n = row0 >> 8;
  const int e0 = row0 & 255;

  v8f acc[4] = {};
  // A-operand: lane l holds row (l&15), K = kb+{0..7} and kb+{16..23}, kb = 8*(l>=16)
  const unsigned short* xh = Xhi + (size_t)(row0 + l15) * 1024 + hi16 * 8;
  const unsigned short* xl = Xlo + (size_t)(row0 + l15) * 1024 + hi16 * 8;
  // B-operand: lane l holds col (l&15), K = kb+{0..7} and kb+{16..23}, kb = 8*(l>=16)
  const size_t wtbase = (size_t)ht * 64 * 1024 + (size_t)hi16 * 8;

  for (int k = 0; k < 1024; k += 32) {
    v16bf ah = ld16(xh + k, 16);
    v16bf al = ld16(xl + k, 16);
#pragma unroll
    for (int nt = 0; nt < 4; ++nt) {
      const unsigned short* ph = Wthi + wtbase + (size_t)(nt * 16 + l15) * 1024 + k;
      const unsigned short* pl = Wtlo + wtbase + (size_t)(nt * 16 + l15) * 1024 + k;
      v16bf bh = ld16(ph, 16);
      v16bf bl = ld16(pl, 16);
      acc[nt] = wmma_bf16(ah, bh, acc[nt]);   // hi*hi
      acc[nt] = wmma_bf16(ah, bl, acc[nt]);   // hi*lo
      acc[nt] = wmma_bf16(al, bh, acc[nt]);   // lo*hi
    }
  }

  // ---- epilogue: s1/s2 row dots (C layout: VGPR v -> row v+8*hi16, col nt*16+l15)
  const float* w1p = w1 + ((size_t)ht * 64 + n) * 64;
  const float* w2p = w2 + ((size_t)ht * 64 + n) * 64;
  float w1v[4], w2v[4];
#pragma unroll
  for (int nt = 0; nt < 4; ++nt) {
    w1v[nt] = w1p[nt * 16 + l15];
    w2v[nt] = w2p[nt * 16 + l15];
  }
  float p1[8], p2[8];
#pragma unroll
  for (int v = 0; v < 8; ++v) {
    p1[v] = acc[0][v] * w1v[0] + acc[1][v] * w1v[1] + acc[2][v] * w1v[2] + acc[3][v] * w1v[3];
    p2[v] = acc[0][v] * w2v[0] + acc[1][v] * w2v[1] + acc[2][v] * w2v[2] + acc[3][v] * w2v[3];
  }
#pragma unroll
  for (int m = 1; m <= 8; m <<= 1) {   // reduce within each 16-lane half
#pragma unroll
    for (int v = 0; v < 8; ++v) {
      p1[v] += __shfl_xor(p1[v], m, 32);
      p2[v] += __shfl_xor(p2[v], m, 32);
    }
  }
  if (l15 == 0) {
    size_t sb = ((size_t)ht * 64 + n) * 256 + e0 + hi16 * 8;
#pragma unroll
    for (int v = 0; v < 8; ++v) {
      s1[sb + v] = p1[v];
      s2[sb + v] = p2[v];
    }
  }
  if (t == 2) {  // store adj_mask-scaled h_last for the attention output GEMM
    const float* am = adj_mask + (size_t)n * 256 + e0 + hi16 * 8;
    size_t hb = (((size_t)h * 64 + n) * 256 + e0 + hi16 * 8) * 64 + l15;
#pragma unroll
    for (int v = 0; v < 8; ++v) {
      float mv = am[v];
#pragma unroll
      for (int nt = 0; nt < 4; ++nt)
        hmask[hb + (size_t)v * 64 + nt * 16] = acc[nt][v] * mv;
    }
  }
}

// ---------------------------------------------------------------------------
// Kernel 5: attention.  One block (256 thr / 8 waves) per (h,n).
// Pass 1: row max + expsum.  Pass 2: regenerate coef chunks of 32 i-rows into
// padded/transposed LDS (bf16 hi/lo), WMMA coefs^T @ h_last, relu, store.
// ---------------------------------------------------------------------------
static __device__ __forceinline__ float attn_score(int t, const float* s1F,
                                                   const float* s2F, int i, int j) {
  if (t == 0) return NEG_INF_F;
  int o = (t - 1) * 256;
  float x = s1F[o + i] + s2F[o + j];
  return x >= 0.f ? x : 0.2f * x;
}

#define CT_STRIDE 40  // bf16 elems per j row (padded: 80B, 16B-aligned, conflict-free)

__global__ __launch_bounds__(256) void attn_kernel(
    const int* __restrict__ adj, const float* __restrict__ s1,
    const float* __restrict__ s2, const float* __restrict__ hmask,
    float* __restrict__ out) {
  __shared__ float s1F[3 * 256];
  __shared__ float s2F[3 * 256];
  __shared__ float rmax[256];
  __shared__ float rinv[256];
  __shared__ unsigned short ctH[256 * CT_STRIDE];
  __shared__ unsigned short ctL[256 * CT_STRIDE];

  const int bid = blockIdx.x;
  const int h = bid & 7;
  const int n = bid >> 3;
  const int tid = threadIdx.x;
  const int wave = tid >> 5;
  const int lane = tid & 31;
  const int l15 = lane & 15;
  const int hi16 = lane >> 4;

#pragma unroll
  for (int t = 0; t < 3; ++t) {
    size_t b = ((size_t)(h * 3 + t) * 64 + n) * 256;
    s1F[t * 256 + tid] = s1[b + tid];
    s2F[t * 256 + tid] = s2[b + tid];
  }
  __syncthreads();

  const int* adjn = adj + (size_t)n * 256 * 256;

  // ---- pass 1: per-row softmax stats (wave w owns rows 32w..32w+31)
  for (int r = 0; r < 32; ++r) {
    int i = wave * 32 + r;
    const int* arow = adjn + i * 256;
    int4 aa = *(const int4*)(arow + 4 * lane);
    int4 ab = *(const int4*)(arow + 128 + 4 * lane);
    float sc0 = attn_score(aa.x, s1F, s2F, i, 4 * lane + 0);
    float sc1 = attn_score(aa.y, s1F, s2F, i, 4 * lane + 1);
    float sc2 = attn_score(aa.z, s1F, s2F, i, 4 * lane + 2);
    float sc3 = attn_score(aa.w, s1F, s2F, i, 4 * lane + 3);
    float sc4 = attn_score(ab.x, s1F, s2F, i, 128 + 4 * lane + 0);
    float sc5 = attn_score(ab.y, s1F, s2F, i, 128 + 4 * lane + 1);
    float sc6 = attn_score(ab.z, s1F, s2F, i, 128 + 4 * lane + 2);
    float sc7 = attn_score(ab.w, s1F, s2F, i, 128 + 4 * lane + 3);
    float m = fmaxf(fmaxf(fmaxf(sc0, sc1), fmaxf(sc2, sc3)),
                    fmaxf(fmaxf(sc4, sc5), fmaxf(sc6, sc7)));
#pragma unroll
    for (int d = 1; d < 32; d <<= 1) m = fmaxf(m, __shfl_xor(m, d, 32));
    float s = __expf(sc0 - m) + __expf(sc1 - m) + __expf(sc2 - m) + __expf(sc3 - m) +
              __expf(sc4 - m) + __expf(sc5 - m) + __expf(sc6 - m) + __expf(sc7 - m);
#pragma unroll
    for (int d = 1; d < 32; d <<= 1) s += __shfl_xor(s, d, 32);
    if (lane == 0) {
      rmax[i] = m;
      rinv[i] = 1.f / s;
    }
  }
  __syncthreads();

  v8f acc[2][4] = {};
  const float* hbase = hmask + (((size_t)h * 64 + n) * 256) * 64;

  for (int ic = 0; ic < 8; ++ic) {
    int i0 = ic * 32;
    // ---- regenerate coef chunk, transposed into LDS (wave owns 4 rows)
    for (int r = 0; r < 4; ++r) {
      int i = i0 + wave * 4 + r;
      int il = i - i0;
      const int* arow = adjn + i * 256;
      int4 aa = *(const int4*)(arow + 4 * lane);
      int4 ab = *(const int4*)(arow + 128 + 4 * lane);
      float m = rmax[i], inv = rinv[i];
#define EMIT(av, jj)                                              \
  do {                                                            \
    float sc_ = attn_score((av), s1F, s2F, i, (jj));              \
    float c_ = __expf(sc_ - m) * inv;                             \
    unsigned short h_ = f2bf(c_);                                 \
    ctH[(jj) * CT_STRIDE + il] = h_;                              \
    ctL[(jj) * CT_STRIDE + il] = f2bf(c_ - bf2f(h_));             \
  } while (0)
      EMIT(aa.x, 4 * lane + 0);
      EMIT(aa.y, 4 * lane + 1);
      EMIT(aa.z, 4 * lane + 2);
      EMIT(aa.w, 4 * lane + 3);
      EMIT(ab.x, 128 + 4 * lane + 0);
      EMIT(ab.y, 128 + 4 * lane + 1);
      EMIT(ab.z, 128 + 4 * lane + 2);
      EMIT(ab.w, 128 + 4 * lane + 3);
#undef EMIT
    }
    __syncthreads();

    // ---- B operands: h_last rows i0+K, lane col d = 16nt + l15, K = 8*hi16 + kk (+8 if kk>=8)
    v16bf bh[4], bl[4];
#pragma unroll
    for (int nt = 0; nt < 4; ++nt) {
      int d = nt * 16 + l15;
      u16x16 uh, ul;
#pragma unroll
      for (int kk = 0; kk < 16; ++kk) {
        float hv = hbase[(size_t)(i0 + hi16 * 8 + kk + (kk >= 8 ? 8 : 0)) * 64 + d];
        unsigned short hb_ = f2bf(hv);
        uh[kk] = hb_;
        ul[kk] = f2bf(hv - bf2f(hb_));
      }
      bh[nt] = __builtin_bit_cast(v16bf, uh);
      bl[nt] = __builtin_bit_cast(v16bf, ul);
    }

    // ---- A operands (coefs^T from LDS) + WMMA
#pragma unroll
    for (int mt = 0; mt < 2; ++mt) {
      int jrow = wave * 32 + mt * 16 + l15;
      const unsigned short* pH = &ctH[jrow * CT_STRIDE + hi16 * 8];
      const unsigned short* pL = &ctL[jrow * CT_STRIDE + hi16 * 8];
      v16bf ah = ld16(pH, 16);
      v16bf al = ld16(pL, 16);
#pragma unroll
      for (int nt = 0; nt < 4; ++nt) {
        acc[mt][nt] = wmma_bf16(ah, bh[nt], acc[mt][nt]);
        acc[mt][nt] = wmma_bf16(ah, bl[nt], acc[mt][nt]);
        acc[mt][nt] = wmma_bf16(al, bh[nt], acc[mt][nt]);
      }
    }
    __syncthreads();
  }

  // ---- epilogue: relu, out[n][j][h*64+d]
#pragma unroll
  for (int mt = 0; mt < 2; ++mt) {
    int j0 = wave * 32 + mt * 16 + hi16 * 8;
#pragma unroll
    for (int v = 0; v < 8; ++v) {
      int j = j0 + v;
      float* op = out + ((size_t)(n * 256 + j)) * 512 + h * 64 + l15;
#pragma unroll
      for (int nt = 0; nt < 4; ++nt)
        op[nt * 16] = fmaxf(acc[mt][nt][v], 0.f);
    }
  }
}

// ---------------------------------------------------------------------------
// Host launcher
// ---------------------------------------------------------------------------
extern "C" void kernel_launch(void* const* d_in, const int* in_sizes, int n_in,
                              void* d_out, int out_size, void* d_ws, size_t ws_size,
                              hipStream_t stream) {
  (void)in_sizes; (void)n_in; (void)out_size; (void)ws_size;
  const float* input_state = (const float*)d_in[0];
  const int*   adj         = (const int*)d_in[1];
  // d_in[2] = entity_mask (unused by the reference computation)
  const float* adj_mask    = (const float*)d_in[3];
  const float* query_vec   = (const float*)d_in[4];
  const float* W           = (const float*)d_in[5];
  const float* a1          = (const float*)d_in[6];
  const float* a2          = (const float*)d_in[7];
  const float* qW1         = (const float*)d_in[8];
  const float* qW2         = (const float*)d_in[9];
  float* out = (float*)d_out;

  char* ws = (char*)d_ws;
  size_t off = 0;
  unsigned short* Xhi  = (unsigned short*)(ws + off); off += (size_t)16384 * 1024 * 2;  // 32 MB
  unsigned short* Xlo  = (unsigned short*)(ws + off); off += (size_t)16384 * 1024 * 2;  // 32 MB
  unsigned short* Wthi = (unsigned short*)(ws + off); off += (size_t)24 * 64 * 1024 * 2; // 3 MB
  unsigned short* Wtlo = (unsigned short*)(ws + off); off += (size_t)24 * 64 * 1024 * 2; // 3 MB
  float* w1 = (float*)(ws + off); off += (size_t)24 * 64 * 64 * 4;    // 384 KB
  float* w2 = (float*)(ws + off); off += (size_t)24 * 64 * 64 * 4;    // 384 KB
  float* s1 = (float*)(ws + off); off += (size_t)24 * 64 * 256 * 4;   // 1.5 MB
  float* s2 = (float*)(ws + off); off += (size_t)24 * 64 * 256 * 4;   // 1.5 MB
  float* hm = (float*)(ws + off); off += (size_t)8 * 64 * 256 * 64 * 4; // 32 MB
  // total workspace: ~106 MB

  prep_x_kernel<<<16384, 256, 0, stream>>>(input_state, Xhi, Xlo);
  prep_w_kernel<<<6144, 256, 0, stream>>>(W, Wthi, Wtlo);
  gate_kernel<<<24, 256, 0, stream>>>(query_vec, qW1, qW2, a1, a2, w1, w2);
  gemm_h_kernel<<<dim3(128, 24), 256, 0, stream>>>(Xhi, Xlo, Wthi, Wtlo, w1, w2,
                                                   adj_mask, s1, s2, hm);
  attn_kernel<<<512, 256, 0, stream>>>(adj, s1, s2, hm, out);
}
